// Qwen3_5MoeTransformerDecoder_50242527429150
// MI455X (gfx1250) — matrix operand order repaired and checked by reading, not compiled, not run on hardware
//
#include <hip/hip_runtime.h>
#include <cstdint>

// ---------------------------------------------------------------------------
// Qwen3.5-MoE decoder FFN block for MI455X (gfx1250, wave32, WMMA)
// B=4 S=1024 H=1024 MI=512 E=16 K=2 SI=2048  -> T=4096 tokens
// Round 3: fix async-to-LDS builtin pointer types (int4 AS1 / int4 AS3).
// ---------------------------------------------------------------------------

typedef __bf16 bf16;
typedef __attribute__((ext_vector_type(16))) __bf16 v16bf;
typedef __attribute__((ext_vector_type(8)))  __bf16 v8bf;
typedef __attribute__((ext_vector_type(8)))  float  v8f;
typedef __attribute__((ext_vector_type(4)))  int    v4i;

static constexpr int T_TOK = 4096;
static constexpr int HD    = 1024;
static constexpr int NE    = 16;
static constexpr int MID   = 512;
static constexpr int SID   = 2048;

static constexpr int AST = 40;  // A-tile LDS row stride (elems): 80B -> <=2-way banks
static constexpr int BST = 40;  // B-tile LDS col stride (elems)

// ---------------- async global->LDS (CDNA5 ASYNCcnt path) -------------------

#if __has_builtin(__builtin_amdgcn_global_load_async_to_lds_b128)
#define HAVE_ASYNC_LDS 1
#else
#define HAVE_ASYNC_LDS 0
#endif

#if HAVE_ASYNC_LDS
typedef __attribute__((address_space(1))) v4i* gvp_t;  // global int4*
typedef __attribute__((address_space(3))) v4i* lvp_t;  // LDS int4*
#endif

__device__ __forceinline__ void async_cp16(const bf16* g, bf16* l) {
#if HAVE_ASYNC_LDS
  __builtin_amdgcn_global_load_async_to_lds_b128((gvp_t)(void*)g, (lvp_t)(void*)l,
                                                 0, 0);
#else
  *(uint4*)l = *(const uint4*)g;
#endif
}

__device__ __forceinline__ void wait_async() {
#if HAVE_ASYNC_LDS
#if __has_builtin(__builtin_amdgcn_s_wait_asynccnt)
  __builtin_amdgcn_s_wait_asynccnt(0);
#else
  asm volatile("s_wait_asynccnt 0x0" ::: "memory");
#endif
#endif
}

// ---------------- WMMA fragment helpers (layouts per CDNA5 ISA 7.12.2) -----

// A (16x32 bf16): lane<16 holds row M=lane, K={0..7,16..23}; lane>=16 holds
// row M=lane-16, K={8..15,24..31}.  As16 is a [16][AST] LDS block.
__device__ __forceinline__ v16bf afrag(const bf16* As16) {
  int lane = threadIdx.x & 31;
  int row  = lane & 15;
  int koff = (lane & 16) ? 8 : 0;
  const bf16* p = As16 + row * AST + koff;
  union { v16bf v; v8bf h[2]; } u;
  u.h[0] = *(const v8bf*)p;        // K chunk 0..7 (or 8..15)
  u.h[1] = *(const v8bf*)(p + 16); // K chunk 16..23 (or 24..31)
  return u.v;
}

// B (32x16 bf16): lane<16 holds col N=lane, K={0..7,16..23}; lane>=16 col N=lane-16,
// K={8..15,24..31}.  Bs16 is a column-major [16 cols][BST] LDS block.
__device__ __forceinline__ v16bf bfrag(const bf16* Bs16) {
  int lane = threadIdx.x & 31;
  int col  = lane & 15;
  int kh   = (lane & 16) ? 8 : 0;
  const bf16* p = Bs16 + col * BST + kh;
  union { v16bf v; v8bf h[2]; } u;
  u.h[0] = *(const v8bf*)p;
  u.h[1] = *(const v8bf*)(p + 16);
  return u.v;
}

__device__ __forceinline__ v8f wmma_bf16(v16bf a, v16bf b, v8f c) {
  return __builtin_amdgcn_wmma_f32_16x16x32_bf16(false, a, false, b,
                                                 (short)0, c, false, false);
}

// ---------------- LDS tile loaders -----------------------------------------

// A tile: 128 rows x 32 K (bf16), rows gathered via srow[] element offsets.
// 256 threads x 32B each; async DMA straight into LDS when available.
__device__ __forceinline__ void load_a_tile(bf16* As, const bf16* __restrict__ base,
                                            const int* srow, int kt) {
  int tid = threadIdx.x;
  int row = tid >> 1;
  int off = (tid & 1) << 4;
  const bf16* src = base + (size_t)srow[row] + kt + off;
  bf16* dst = As + row * AST + off;
  async_cp16(src, dst);
  async_cp16(src + 8, dst + 8);
}

// B tile: 32 K x 64 N from fp32 weights (row-major, leading dim ldb),
// converted to bf16 and stored COLUMN-major [64][BST] in LDS.
__device__ __forceinline__ void load_b_tile(bf16* Bs, const float* __restrict__ W,
                                            int ldb) {
  int tid = threadIdx.x;
  int col = tid & 63;
  int k0  = (tid >> 6) << 3;
#pragma unroll
  for (int i = 0; i < 8; ++i)
    Bs[col * BST + k0 + i] = (bf16)W[(size_t)(k0 + i) * ldb + col];
}

// ---------------- Kernel 1: router + x cast + shared gate scalar -----------

__global__ __launch_bounds__(256) void k_router(
    const float* __restrict__ x, const float* __restrict__ rw,
    const float* __restrict__ segw, float* __restrict__ logits,
    bf16* __restrict__ xb, float* __restrict__ sgate,
    float* __restrict__ topw, int* __restrict__ topi) {
  __shared__ float xs[8][HD];
  int wid = threadIdx.x >> 5, lane = threadIdx.x & 31;
  int t = blockIdx.x * 8 + wid;
  float* xr = xs[wid];
  for (int i = lane; i < HD; i += 32) {
    float v = x[(size_t)t * HD + i];
    xr[i] = v;
    xb[(size_t)t * HD + i] = (bf16)v;
  }
  __syncthreads();

  // shared-expert gate scalar: sigmoid(x . shared_expert_gate_w)
  float acc = 0.f;
  for (int i = lane; i < HD; i += 32) acc += xr[i] * segw[i];
#pragma unroll
  for (int m = 16; m >= 1; m >>= 1) acc += __shfl_xor(acc, m, 32);
  if (lane == 0) sgate[t] = 1.f / (1.f + __expf(-acc));

  // router logits: lanes 0..15, one expert each
  int e = lane & 15;
  float le = -3.4e38f;
  if (lane < 16) {
    float a = 0.f;
    for (int k2 = 0; k2 < HD; ++k2) a += xr[k2] * rw[(size_t)k2 * NE + e];
    le = a;
    logits[(size_t)t * NE + e] = a;
  }
  // softmax over the 16 lower lanes (butterflies with mask<=8 stay in-group)
  float mx = le;
#pragma unroll
  for (int m = 8; m >= 1; m >>= 1) mx = fmaxf(mx, __shfl_xor(mx, m, 32));
  float pe = (lane < 16) ? __expf(le - mx) : 0.f;
  float s = pe;
#pragma unroll
  for (int m = 8; m >= 1; m >>= 1) s += __shfl_xor(s, m, 32);
  float p = pe / s;
  // top-1 (value, index) with first-occurrence tie break
  float p1 = p; int i1 = e;
#pragma unroll
  for (int m = 8; m >= 1; m >>= 1) {
    float op = __shfl_xor(p1, m, 32);
    int   oi = __shfl_xor(i1, m, 32);
    if (op > p1 || (op == p1 && oi < i1)) { p1 = op; i1 = oi; }
  }
  // top-2
  float q = (e == i1) ? -1.f : p;
  float p2 = q; int i2 = e;
#pragma unroll
  for (int m = 8; m >= 1; m >>= 1) {
    float op = __shfl_xor(p2, m, 32);
    int   oi = __shfl_xor(i2, m, 32);
    if (op > p2 || (op == p2 && oi < i2)) { p2 = op; i2 = oi; }
  }
  if (lane == 0) {
    float inv = 1.f / (p1 + p2);
    topw[2 * t]     = p1 * inv;
    topw[2 * t + 1] = p2 * inv;
    topi[2 * t]     = i1;
    topi[2 * t + 1] = i2;
  }
}

// ---------------- Kernel 2/3: expert assignment lists ----------------------

__global__ void k_zero(int* counts) {
  if (threadIdx.x < NE) counts[threadIdx.x] = 0;
}

__global__ __launch_bounds__(256) void k_assign(const int* __restrict__ topi,
                                                int* counts, int* entries) {
  int t = blockIdx.x * 256 + threadIdx.x;
#pragma unroll
  for (int k = 0; k < 2; ++k) {
    int e = topi[2 * t + k];
    int i = atomicAdd(&counts[e], 1);
    entries[(size_t)e * T_TOK + i] = 2 * t + k;  // encodes (token, slot)
  }
}

// ---------------- Kernel 4: shared expert gate+up GEMM + SwiGLU ------------
// hsh[T][SI] (bf16) = up * silu(gate),  M=T N=SI K=H

__global__ __launch_bounds__(256) void k_shared_gu(
    const bf16* __restrict__ xb, const float* __restrict__ wg,
    const float* __restrict__ wu, bf16* __restrict__ hsh) {
  __shared__ alignas(16) bf16 As[2][128 * AST];
  __shared__ alignas(16) bf16 Bg[2][64 * BST];
  __shared__ alignas(16) bf16 Bu[2][64 * BST];
  __shared__ int srow[128];
  int tid = threadIdx.x, wid = tid >> 5, lane = tid & 31;
  int nb = blockIdx.x * 64, mb = blockIdx.y * 128;
  if (tid < 128) srow[tid] = (mb + tid) * HD;
  __syncthreads();
  v8f ag[4] = {}; v8f au[4] = {};
  const float* wgp = wg + nb;
  const float* wup = wu + nb;
  // prologue: fill buffer 0
  load_a_tile(As[0], xb, srow, 0);
  load_b_tile(Bg[0], wgp, SID);
  load_b_tile(Bu[0], wup, SID);
  constexpr int NIT = HD / 32;
  for (int it = 0; it < NIT; ++it) {
    int cur = it & 1, nxt = cur ^ 1;
    wait_async();
    __syncthreads();
    if (it + 1 < NIT) {
      int kt = (it + 1) * 32;
      load_a_tile(As[nxt], xb, srow, kt);
      load_b_tile(Bg[nxt], wgp + (size_t)kt * SID, SID);
      load_b_tile(Bu[nxt], wup + (size_t)kt * SID, SID);
      if (it + 2 < NIT) {
        __builtin_prefetch(wgp + (size_t)(kt + 32) * SID, 0, 1);
        __builtin_prefetch(wup + (size_t)(kt + 32) * SID, 0, 1);
      }
    }
    v16bf a = afrag(As[cur] + wid * 16 * AST);
#pragma unroll
    for (int j = 0; j < 4; ++j) {
      ag[j] = wmma_bf16(a, bfrag(Bg[cur] + j * 16 * BST), ag[j]);
      au[j] = wmma_bf16(a, bfrag(Bu[cur] + j * 16 * BST), au[j]);
    }
  }
  int rbase = mb + wid * 16 + ((lane & 16) >> 1);
  int cbase = nb + (lane & 15);
#pragma unroll
  for (int r = 0; r < 8; ++r)
#pragma unroll
    for (int j = 0; j < 4; ++j) {
      float g = ag[j][r], u = au[j][r];
      float h = u * g / (1.f + __expf(-g));  // u * silu(g)
      hsh[(size_t)(rbase + r) * SID + cbase + j * 16] = (bf16)h;
    }
}

// ---------------- Kernel 5: shared expert down GEMM + sigmoid gating -------
// out[T][H] = sgate[t] * (hsh @ wsd),  M=T N=H K=SI

__global__ __launch_bounds__(256) void k_shared_down(
    const bf16* __restrict__ hsh, const float* __restrict__ wd,
    const float* __restrict__ sgate, float* __restrict__ out) {
  __shared__ alignas(16) bf16 As[2][128 * AST];
  __shared__ alignas(16) bf16 Bs[2][64 * BST];
  __shared__ int srow[128];
  int tid = threadIdx.x, wid = tid >> 5, lane = tid & 31;
  int nb = blockIdx.x * 64, mb = blockIdx.y * 128;
  if (tid < 128) srow[tid] = (mb + tid) * SID;
  __syncthreads();
  v8f acc[4] = {};
  const float* wdp = wd + nb;
  load_a_tile(As[0], hsh, srow, 0);
  load_b_tile(Bs[0], wdp, HD);
  constexpr int NIT = SID / 32;
  for (int it = 0; it < NIT; ++it) {
    int cur = it & 1, nxt = cur ^ 1;
    wait_async();
    __syncthreads();
    if (it + 1 < NIT) {
      int kt = (it + 1) * 32;
      load_a_tile(As[nxt], hsh, srow, kt);
      load_b_tile(Bs[nxt], wdp + (size_t)kt * HD, HD);
      if (it + 2 < NIT) __builtin_prefetch(wdp + (size_t)(kt + 32) * HD, 0, 1);
    }
    v16bf a = afrag(As[cur] + wid * 16 * AST);
#pragma unroll
    for (int j = 0; j < 4; ++j)
      acc[j] = wmma_bf16(a, bfrag(Bs[cur] + j * 16 * BST), acc[j]);
  }
  int rbase = mb + wid * 16 + ((lane & 16) >> 1);
  int cbase = nb + (lane & 15);
#pragma unroll
  for (int r = 0; r < 8; ++r) {
    float sg = sgate[rbase + r];
#pragma unroll
    for (int j = 0; j < 4; ++j)
      out[(size_t)(rbase + r) * HD + cbase + j * 16] = sg * acc[j][r];
  }
}

// ---------------- Kernel 6: gathered expert gate+up GEMM + SwiGLU ----------
// hid[2t+k][MI] = up_e * silu(gate_e) for each assignment (t,k) of expert e

__global__ __launch_bounds__(256) void k_expert_gu(
    const bf16* __restrict__ xb, const float* __restrict__ egu,
    const int* __restrict__ counts, const int* __restrict__ entries,
    bf16* __restrict__ hid) {
  __shared__ alignas(16) bf16 As[2][128 * AST];
  __shared__ alignas(16) bf16 Bg[2][64 * BST];
  __shared__ alignas(16) bf16 Bu[2][64 * BST];
  __shared__ int srow[128];
  __shared__ int sent[128];
  int e = blockIdx.z;
  int cnt = counts[e];
  int mb = blockIdx.y * 128;
  if (mb >= cnt) return;
  int tid = threadIdx.x, wid = tid >> 5, lane = tid & 31;
  int nb = blockIdx.x * 64;  // hidden-dim column block (0..MID)
  if (tid < 128) {
    int i = mb + tid;
    int ent = entries[(size_t)e * T_TOK + (i < cnt ? i : cnt - 1)];
    sent[tid] = ent;
    srow[tid] = (ent >> 1) * HD;  // gather x row of token t = ent>>1
  }
  __syncthreads();
  v8f ag[4] = {}; v8f au[4] = {};
  const float* wgp = egu + (size_t)e * HD * (2 * MID) + nb;
  const float* wup = wgp + MID;
  load_a_tile(As[0], xb, srow, 0);
  load_b_tile(Bg[0], wgp, 2 * MID);
  load_b_tile(Bu[0], wup, 2 * MID);
  constexpr int NIT = HD / 32;
  for (int it = 0; it < NIT; ++it) {
    int cur = it & 1, nxt = cur ^ 1;
    wait_async();
    __syncthreads();
    if (it + 1 < NIT) {
      int kt = (it + 1) * 32;
      load_a_tile(As[nxt], xb, srow, kt);
      load_b_tile(Bg[nxt], wgp + (size_t)kt * (2 * MID), 2 * MID);
      load_b_tile(Bu[nxt], wup + (size_t)kt * (2 * MID), 2 * MID);
      if (it + 2 < NIT) {
        __builtin_prefetch(wgp + (size_t)(kt + 32) * (2 * MID), 0, 1);
        __builtin_prefetch(wup + (size_t)(kt + 32) * (2 * MID), 0, 1);
      }
    }
    v16bf a = afrag(As[cur] + wid * 16 * AST);
#pragma unroll
    for (int j = 0; j < 4; ++j) {
      ag[j] = wmma_bf16(a, bfrag(Bg[cur] + j * 16 * BST), ag[j]);
      au[j] = wmma_bf16(a, bfrag(Bu[cur] + j * 16 * BST), au[j]);
    }
  }
  int cbase = nb + (lane & 15);
#pragma unroll
  for (int r = 0; r < 8; ++r) {
    int rloc = wid * 16 + r + ((lane & 16) >> 1);
    if (mb + rloc < cnt) {
      int ent = sent[rloc];
#pragma unroll
      for (int j = 0; j < 4; ++j) {
        float g = ag[j][r], u = au[j][r];
        float h = u * g / (1.f + __expf(-g));
        hid[(size_t)ent * MID + cbase + j * 16] = (bf16)h;
      }
    }
  }
}

// ---------------- Kernel 7: gathered expert down GEMM + routing weight -----
// contrib[2t+k][H] = topw[2t+k] * (hid[2t+k] @ down_e)

__global__ __launch_bounds__(256) void k_expert_down(
    const bf16* __restrict__ hid, const float* __restrict__ edw,
    const int* __restrict__ counts, const int* __restrict__ entries,
    const float* __restrict__ topw, float* __restrict__ contrib) {
  __shared__ alignas(16) bf16 As[2][128 * AST];
  __shared__ alignas(16) bf16 Bs[2][64 * BST];
  __shared__ int srow[128];
  __shared__ int sent[128];
  __shared__ float swt[128];
  int e = blockIdx.z;
  int cnt = counts[e];
  int mb = blockIdx.y * 128;
  if (mb >= cnt) return;
  int tid = threadIdx.x, wid = tid >> 5, lane = tid & 31;
  int nb = blockIdx.x * 64;  // H column block
  if (tid < 128) {
    int i = mb + tid;
    int ent = entries[(size_t)e * T_TOK + (i < cnt ? i : cnt - 1)];
    sent[tid] = ent;
    srow[tid] = ent * MID;
    swt[tid] = topw[ent];
  }
  __syncthreads();
  v8f acc[4] = {};
  const float* wdp = edw + (size_t)e * MID * HD + nb;
  load_a_tile(As[0], hid, srow, 0);
  load_b_tile(Bs[0], wdp, HD);
  constexpr int NIT = MID / 32;
  for (int it = 0; it < NIT; ++it) {
    int cur = it & 1, nxt = cur ^ 1;
    wait_async();
    __syncthreads();
    if (it + 1 < NIT) {
      int kt = (it + 1) * 32;
      load_a_tile(As[nxt], hid, srow, kt);
      load_b_tile(Bs[nxt], wdp + (size_t)kt * HD, HD);
      if (it + 2 < NIT) __builtin_prefetch(wdp + (size_t)(kt + 32) * HD, 0, 1);
    }
    v16bf a = afrag(As[cur] + wid * 16 * AST);
#pragma unroll
    for (int j = 0; j < 4; ++j)
      acc[j] = wmma_bf16(a, bfrag(Bs[cur] + j * 16 * BST), acc[j]);
  }
  int cbase = nb + (lane & 15);
#pragma unroll
  for (int r = 0; r < 8; ++r) {
    int rloc = wid * 16 + r + ((lane & 16) >> 1);
    if (mb + rloc < cnt) {
      int ent = sent[rloc];
      float wgt = swt[rloc];
#pragma unroll
      for (int j = 0; j < 4; ++j)
        contrib[(size_t)ent * HD + cbase + j * 16] = wgt * acc[j][r];
    }
  }
}

// ---------------- Kernel 8: final accumulate -------------------------------

__global__ __launch_bounds__(256) void k_final(float* __restrict__ out,
                                               const float* __restrict__ contrib) {
  size_t i = (size_t)blockIdx.x * 256 + threadIdx.x;  // i < T*H
  int t = (int)(i >> 10);
  int h = (int)(i & 1023);
  out[i] += contrib[(size_t)(2 * t) * HD + h] + contrib[(size_t)(2 * t + 1) * HD + h];
}

// ---------------------------------------------------------------------------

extern "C" void kernel_launch(void* const* d_in, const int* in_sizes, int n_in,
                              void* d_out, int out_size, void* d_ws, size_t ws_size,
                              hipStream_t stream) {
  (void)in_sizes; (void)n_in; (void)out_size; (void)ws_size;
  const float* x    = (const float*)d_in[0];  // [B,S,H]
  const float* rw   = (const float*)d_in[1];  // [H,E]
  const float* egu  = (const float*)d_in[2];  // [E,H,2*MI]
  const float* edw  = (const float*)d_in[3];  // [E,MI,H]
  const float* wsg  = (const float*)d_in[4];  // [H,SI]
  const float* wsu  = (const float*)d_in[5];  // [H,SI]
  const float* wsd  = (const float*)d_in[6];  // [SI,H]
  const float* segw = (const float*)d_in[7];  // [H,1]
  float* out    = (float*)d_out;
  float* logits = out + (size_t)T_TOK * HD;

  char* w = (char*)d_ws;
  bf16*  xb      = (bf16*)w;  w += (size_t)T_TOK * HD * 2;        // 8 MiB
  bf16*  hsh     = (bf16*)w;  w += (size_t)T_TOK * SID * 2;       // 16 MiB
  bf16*  hid     = (bf16*)w;  w += (size_t)2 * T_TOK * MID * 2;   // 8 MiB
  float* contrib = (float*)w; w += (size_t)2 * T_TOK * HD * 4;    // 32 MiB
  float* sgate   = (float*)w; w += (size_t)T_TOK * 4;
  float* topw    = (float*)w; w += (size_t)2 * T_TOK * 4;
  int*   topi    = (int*)w;   w += (size_t)2 * T_TOK * 4;
  int*   counts  = (int*)w;   w += 256;
  int*   entries = (int*)w;   w += (size_t)NE * T_TOK * 4;        // 256 KiB

  k_router<<<T_TOK / 8, 256, 0, stream>>>(x, rw, segw, logits, xb, sgate, topw, topi);
  k_zero<<<1, 64, 0, stream>>>(counts);
  k_assign<<<T_TOK / 256, 256, 0, stream>>>(topi, counts, entries);
  k_shared_gu<<<dim3(SID / 64, T_TOK / 128), 256, 0, stream>>>(xb, wsg, wsu, hsh);
  k_shared_down<<<dim3(HD / 64, T_TOK / 128), 256, 0, stream>>>(hsh, wsd, sgate, out);
  k_expert_gu<<<dim3(MID / 64, T_TOK / 128, NE), 256, 0, stream>>>(xb, egu, counts, entries, hid);
  k_expert_down<<<dim3(HD / 64, T_TOK / 128, NE), 256, 0, stream>>>(hid, edw, counts, entries, topw, contrib);
  k_final<<<(T_TOK * HD) / 256, 256, 0, stream>>>(out, contrib);
}
